// SelectionGNN_CliqueLine_6090263626210
// MI455X (gfx1250) — compile-verified, matrix operand order checked by reading, not run on hardware
//
#include <hip/hip_runtime.h>
#include <hip/hip_bf16.h>

// ---------------------------------------------------------------------------
// SelectionGNN (clique + line expansion), CDNA5/gfx1250.
// Diffusion GEMMs (Z_out = Z_in * S^T) on v_wmma_f32_16x16x32_f16, fp32 acc.
// All activations kept fp16; S converted to fp16 once per launch (L2-resident).
// LDS double-buffered; stage-in uses GLOBAL_LOAD_ASYNC_TO_LDS_B128 (ASYNCcnt).
// ---------------------------------------------------------------------------

typedef __attribute__((ext_vector_type(16))) _Float16 v16h;
typedef __attribute__((ext_vector_type(8)))  _Float16 v8h;
typedef __attribute__((ext_vector_type(8)))  float    v8f;
typedef int v4i __attribute__((vector_size(16)));   // gcc-style, matches builtin

#define NNODES 4096
#define BATCH  8

#if defined(__AMDGCN__) && __has_builtin(__builtin_amdgcn_global_load_async_to_lds_b128) && __has_builtin(__builtin_amdgcn_s_wait_asynccnt)
#define USE_ASYNC 1
#else
#define USE_ASYNC 0
#endif

#if USE_ASYNC
typedef __attribute__((address_space(1))) v4i* glob4_t;
typedef __attribute__((address_space(3))) v4i* lds4_t;
#endif

// 16-byte global->LDS copy (async when available)
__device__ __forceinline__ void cp16(const _Float16* g, _Float16* l) {
#if USE_ASYNC
    __builtin_amdgcn_global_load_async_to_lds_b128(
        (glob4_t)(void*)(const void*)g, (lds4_t)(void*)l, 0, 0);
#else
    *(v8h*)l = *(const v8h*)g;
#endif
}

template <int N>
__device__ __forceinline__ void wait_async() {
#if USE_ASYNC
    __builtin_amdgcn_s_wait_asynccnt(N);
#endif
}

// ---------------------------------------------------------------------------
// fp32 -> fp16 conversion, 8 elements / thread
// ---------------------------------------------------------------------------
__global__ __launch_bounds__(256) void cvt_f32_f16(const float* __restrict__ src,
                                                   _Float16* __restrict__ dst) {
    size_t i = ((size_t)blockIdx.x * 256 + threadIdx.x) * 8;
    float4 a = *(const float4*)(src + i);
    float4 b = *(const float4*)(src + i + 4);
    v8h o;
    o[0] = (_Float16)a.x; o[1] = (_Float16)a.y; o[2] = (_Float16)a.z; o[3] = (_Float16)a.w;
    o[4] = (_Float16)b.x; o[5] = (_Float16)b.y; o[6] = (_Float16)b.z; o[7] = (_Float16)b.w;
    *(v8h*)(dst + i) = o;
}

// ---------------------------------------------------------------------------
// Diffusion GEMM:  C[r][n] = sum_m A[r][m] * S[n][m]   (C = A * S^T), all fp16
//   Ah: fp16 [R][4096], Sh: fp16 [4096][4096], Ch: fp16 [R][4096]
// WG = 256 threads (8 waves). Tile: 128 rows x 128 cols, K-step 32.
// Wave w owns row-tile w (16 rows), loops 8 col-tiles; LDS double-buffered.
// ---------------------------------------------------------------------------
#define LDS_STRIDE 40   // halfs per row (padded: 80B = 20 banks, conflict-light)

__global__ __launch_bounds__(256) void diffuse_wmma(const _Float16* __restrict__ Ah,
                                                    const _Float16* __restrict__ Sh,
                                                    _Float16* __restrict__ Ch) {
    __shared__ _Float16 lA[2][128][LDS_STRIDE];
    __shared__ _Float16 lS[2][128][LDS_STRIDE];

    const int tid  = threadIdx.x;
    const int wave = tid >> 5;
    const int lane = tid & 31;
    const int lr   = lane & 15;            // row/col within 16-wide fragment
    const int h8   = (lane >> 4) * 8;      // K half-offset (0 or 8) per WMMA layout
    const int n0   = blockIdx.x * 128;     // output-node tile base
    const int r0   = blockIdx.y * 128;     // row tile base

    // per-thread staging slots: tile is 128 rows x 32 halfs = 4096 halfs;
    // each thread moves 2x8 halfs of A and 2x8 halfs of S (4 b128 ops).
    const int e0  = tid * 8;
    const int e1  = e0 + 2048;
    const int ra0 = e0 >> 5, ca0 = e0 & 31;
    const int ra1 = e1 >> 5, ca1 = e1 & 31;

    v8f acc[8];
#pragma unroll
    for (int c = 0; c < 8; ++c)
        acc[c] = (v8f){0.f, 0.f, 0.f, 0.f, 0.f, 0.f, 0.f, 0.f};

    union V16 { v16h v; v8h h[2]; };

    auto stage = [&](int buf, int k0) {
        cp16(Ah + (size_t)(r0 + ra0) * NNODES + k0 + ca0, &lA[buf][ra0][ca0]);
        cp16(Ah + (size_t)(r0 + ra1) * NNODES + k0 + ca1, &lA[buf][ra1][ca1]);
        cp16(Sh + (size_t)(n0 + ra0) * NNODES + k0 + ca0, &lS[buf][ra0][ca0]);
        cp16(Sh + (size_t)(n0 + ra1) * NNODES + k0 + ca1, &lS[buf][ra1][ca1]);
    };

    stage(0, 0);  // prologue: tile 0 in flight

    for (int it = 0; it < NNODES / 32; ++it) {
        const int cur = it & 1;
        if (it < NNODES / 32 - 1) {
            stage(cur ^ 1, (it + 1) * 32);  // prefetch next tile
            wait_async<4>();                // newest 4 = next tile; current done
        } else {
            wait_async<0>();
        }
        __syncthreads();

        V16 afr;
        afr.h[0] = *(const v8h*)&lA[cur][wave * 16 + lr][h8];
        afr.h[1] = *(const v8h*)&lA[cur][wave * 16 + lr][h8 + 16];
#pragma unroll
        for (int c = 0; c < 8; ++c) {
            V16 bfr;  // B[k][n] = S[n][k]: lane column n reads row n of S tile
            bfr.h[0] = *(const v8h*)&lS[cur][c * 16 + lr][h8];
            bfr.h[1] = *(const v8h*)&lS[cur][c * 16 + lr][h8 + 16];
            acc[c] = __builtin_amdgcn_wmma_f32_16x16x32_f16(
                false, afr.v, false, bfr.v, (short)0, acc[c], false, false);
        }
        __syncthreads();
    }

    // write C (fp16) per documented C/D layout
#pragma unroll
    for (int c = 0; c < 8; ++c) {
#pragma unroll
        for (int j = 0; j < 8; ++j) {
            int row = r0 + wave * 16 + j + h8;   // M = j (lanes 0-15) or j+8
            int col = n0 + c * 16 + lr;
            Ch[(size_t)row * NNODES + col] = (_Float16)acc[c][j];
        }
    }
}

// ---------------------------------------------------------------------------
// Tap combination: y[b,g,n] = relu(bias[g] + sum_{k,f} h[g,0,k,f] * z_k[b,f,n])
// fp16 taps in, f32 accumulate, fp16 out. One thread per (b, n).
// ---------------------------------------------------------------------------
template <int FIN, int FOUT>
__global__ __launch_bounds__(256) void combine_taps(const _Float16* __restrict__ z0,
                                                    const _Float16* __restrict__ z1,
                                                    const _Float16* __restrict__ z2,
                                                    const float* __restrict__ h,
                                                    const float* __restrict__ bias,
                                                    _Float16* __restrict__ y) {
    int t = blockIdx.x * 256 + threadIdx.x;
    int b = t >> 12;          // / 4096
    int n = t & (NNODES - 1);

    const _Float16* zs[3] = {z0, z1, z2};
    float z[3][FIN];
#pragma unroll
    for (int k = 0; k < 3; ++k)
#pragma unroll
        for (int f = 0; f < FIN; ++f)
            z[k][f] = (float)zs[k][((size_t)b * FIN + f) * NNODES + n];

    for (int g = 0; g < FOUT; ++g) {
        float acc = bias[g];
#pragma unroll
        for (int k = 0; k < 3; ++k)
#pragma unroll
            for (int f = 0; f < FIN; ++f)
                acc = fmaf(h[(g * 3 + k) * FIN + f], z[k][f], acc);
        y[((size_t)b * FOUT + g) * NNODES + n] = (_Float16)fmaxf(acc, 0.f);
    }
}

// ---------------------------------------------------------------------------
// MLP head, stage 1: H[b][j] = relu(dot(y[b,:], Wm1[j,:]) + bm1[j])
// y fp16 (65536 per batch), Wm1 fp32; block per (b, j); LDS tree reduce.
// ---------------------------------------------------------------------------
__global__ __launch_bounds__(256) void mlp_hidden(const _Float16* __restrict__ y,
                                                  const float* __restrict__ Wm1,
                                                  const float* __restrict__ bm1,
                                                  float* __restrict__ H) {
    const int b = blockIdx.x >> 6;
    const int j = blockIdx.x & 63;
    const v8h*    y8 = (const v8h*)(y + (size_t)b * 65536);
    const float4* w4 = (const float4*)(Wm1 + (size_t)j * 65536);
    float s = 0.f;
    for (int i = threadIdx.x; i < 8192; i += 256) {
        v8h a = y8[i];
        float4 w0 = w4[2 * i], w1 = w4[2 * i + 1];
        s += (float)a[0] * w0.x + (float)a[1] * w0.y + (float)a[2] * w0.z + (float)a[3] * w0.w
           + (float)a[4] * w1.x + (float)a[5] * w1.y + (float)a[6] * w1.z + (float)a[7] * w1.w;
    }
    __shared__ float red[256];
    red[threadIdx.x] = s;
    __syncthreads();
    for (int off = 128; off > 0; off >>= 1) {
        if (threadIdx.x < off) red[threadIdx.x] += red[threadIdx.x + off];
        __syncthreads();
    }
    if (threadIdx.x == 0) H[b * 64 + j] = fmaxf(red[0] + bm1[j], 0.f);
}

// ---------------------------------------------------------------------------
// MLP head, stage 2: out[b] = dot(H[b,:], Wm2[0,:]) + bm2[0]
// ---------------------------------------------------------------------------
__global__ __launch_bounds__(64) void mlp_out(const float* __restrict__ H,
                                              const float* __restrict__ Wm2,
                                              const float* __restrict__ bm2,
                                              float* __restrict__ out) {
    __shared__ float red[64];
    int j = threadIdx.x;
    for (int b = 0; b < BATCH; ++b) {
        red[j] = H[b * 64 + j] * Wm2[j];
        __syncthreads();
        for (int off = 32; off > 0; off >>= 1) {
            if (j < off) red[j] += red[j + off];
            __syncthreads();
        }
        if (j == 0) out[b] = red[0] + bm2[0];
        __syncthreads();
    }
}

// ---------------------------------------------------------------------------
// Host orchestration
// ---------------------------------------------------------------------------
extern "C" void kernel_launch(void* const* d_in, const int* in_sizes, int n_in,
                              void* d_out, int out_size, void* d_ws, size_t ws_size,
                              hipStream_t stream) {
    const float* x   = (const float*)d_in[0];
    const float* Sc  = (const float*)d_in[1];
    const float* Sl  = (const float*)d_in[2];
    const float* Wc1 = (const float*)d_in[3];
    const float* bc1 = (const float*)d_in[4];
    const float* Wc2 = (const float*)d_in[5];
    const float* bc2 = (const float*)d_in[6];
    const float* Wl1 = (const float*)d_in[7];
    const float* bl1 = (const float*)d_in[8];
    const float* Wl2 = (const float*)d_in[9];
    const float* bl2 = (const float*)d_in[10];
    const float* Wm1 = (const float*)d_in[11];
    const float* bm1 = (const float*)d_in[12];
    const float* Wm2 = (const float*)d_in[13];
    const float* bm2 = (const float*)d_in[14];
    float* out = (float*)d_out;

    const size_t S_ELEMS  = (size_t)NNODES * NNODES;           // 16.78M
    const size_t S_HBYTES = S_ELEMS * sizeof(_Float16);        // 33.55 MB
    const size_t XH_B     = (size_t)BATCH * 16 * NNODES * sizeof(_Float16); // 1 MB
    const size_t BUF_B    = (size_t)BATCH * 32 * NNODES * sizeof(_Float16); // 2 MB

    char* ws = (char*)d_ws;
    _Float16* Sch = (_Float16*)(ws);
    _Float16* Slh = (_Float16*)(ws + S_HBYTES);
    _Float16* xh  = (_Float16*)(ws + 2 * S_HBYTES);
    _Float16* tA  = (_Float16*)(ws + 2 * S_HBYTES + XH_B);
    _Float16* tB  = (_Float16*)(ws + 2 * S_HBYTES + XH_B + 1 * BUF_B);
    _Float16* yC  = (_Float16*)(ws + 2 * S_HBYTES + XH_B + 2 * BUF_B);
    _Float16* yD  = (_Float16*)(ws + 2 * S_HBYTES + XH_B + 3 * BUF_B);
    float*    Hb  = (float*)   (ws + 2 * S_HBYTES + XH_B + 4 * BUF_B);

    cvt_f32_f16<<<(int)(S_ELEMS / 2048), 256, 0, stream>>>(Sc, Sch);
    cvt_f32_f16<<<(int)(S_ELEMS / 2048), 256, 0, stream>>>(Sl, Slh);
    cvt_f32_f16<<<(int)((BATCH * 16 * NNODES) / 2048), 256, 0, stream>>>(x, xh);

    const dim3 g128(NNODES / 128, 1);   // R = 8*16 = 128 rows
    const dim3 g256(NNODES / 128, 2);   // R = 8*32 = 256 rows
    const int cgrid = BATCH * NNODES / 256;

    // ---- clique layer 1: Fin=16 -> Fout=32 ----
    diffuse_wmma<<<g128, 256, 0, stream>>>(xh, Sch, tA);
    diffuse_wmma<<<g128, 256, 0, stream>>>(tA, Sch, tB);
    combine_taps<16, 32><<<cgrid, 256, 0, stream>>>(xh, tA, tB, Wc1, bc1, yC);

    // ---- clique layer 2: Fin=32 -> Fout=16 ----
    diffuse_wmma<<<g256, 256, 0, stream>>>(yC, Sch, tA);
    diffuse_wmma<<<g256, 256, 0, stream>>>(tA, Sch, tB);
    combine_taps<32, 16><<<cgrid, 256, 0, stream>>>(yC, tA, tB, Wc2, bc2, yD);

    // ---- line layer 1: Fin=16 -> Fout=32 ----
    diffuse_wmma<<<g128, 256, 0, stream>>>(yD, Slh, tA);
    diffuse_wmma<<<g128, 256, 0, stream>>>(tA, Slh, tB);
    combine_taps<16, 32><<<cgrid, 256, 0, stream>>>(yD, tA, tB, Wl1, bl1, yC);

    // ---- line layer 2: Fin=32 -> Fout=16 ----
    diffuse_wmma<<<g256, 256, 0, stream>>>(yC, Slh, tA);
    diffuse_wmma<<<g256, 256, 0, stream>>>(tA, Slh, tB);
    combine_taps<32, 16><<<cgrid, 256, 0, stream>>>(yC, tA, tB, Wl2, bl2, yD);

    // ---- MLP head ----
    mlp_hidden<<<BATCH * 64, 256, 0, stream>>>(yD, Wm1, bm1, Hb);
    mlp_out<<<1, 64, 0, stream>>>(Hb, Wm2, bm2, out);
}
